// SAGE_15384572854544
// MI455X (gfx1250) — compile-verified, matrix operand order checked by reading, not run on hardware
//
#include <hip/hip_runtime.h>
#include <math.h>

// ---------------------------------------------------------------------------
// GraphSAGE forward for MI455X (gfx1250, wave32).
//   h   = x @ W_pre + b_pre
//   a1  = mean-aggregate(h) ; h2 = relu(a1 @ W1 + b1)
//   a2  = mean-aggregate(h2); out = a2 @ W2 + b2 ; row-L2-normalize
// GEMMs use v_wmma_f32_16x16x32_f16 (f16 in / f32 accumulate).
// Aggregation is L2-resident float4 gather + global_atomic_add_f32 scatter.
// ---------------------------------------------------------------------------

typedef __attribute__((ext_vector_type(16))) _Float16 v16h;
typedef __attribute__((ext_vector_type(8)))  _Float16 v8h;
typedef __attribute__((ext_vector_type(8)))  float    v8f;

#define DK 128   // inner dim of every GEMM (D_IN = D_FEAT = D_HID = 128)

// ---------------- small utility kernels ----------------

__global__ void sage_fill_kernel(float* __restrict__ p, float val, int n) {
    int i = blockIdx.x * blockDim.x + threadIdx.x;
    if (i < n) p[i] = val;
}

// cnt[dst] += 1 per edge (cnt pre-initialized to 1.0 for the self-loop)
__global__ void sage_degree_kernel(const int* __restrict__ dst,
                                   float* __restrict__ cnt, int E) {
    int e = blockIdx.x * blockDim.x + threadIdx.x;
    if (e < E) atomicAdd(&cnt[dst[e]], 1.0f);
}

// One wave per edge; lane covers 4 consecutive features (128 = 32 lanes * 4).
__global__ void sage_scatter_kernel(const float* __restrict__ h,
                                    const int* __restrict__ src,
                                    const int* __restrict__ dst,
                                    float* __restrict__ sum, int E) {
    int t    = blockIdx.x * blockDim.x + threadIdx.x;
    int e    = t >> 5;
    int lane = t & 31;
    if (e >= E) return;
    int s = src[e];
    int d = dst[e];
    float4 v = reinterpret_cast<const float4*>(h + (size_t)s * DK)[lane];
    float* o = sum + (size_t)d * DK + lane * 4;
    atomicAdd(o + 0, v.x);
    atomicAdd(o + 1, v.y);
    atomicAdd(o + 2, v.z);
    atomicAdd(o + 3, v.w);
}

// agg[n][f] = (sum[n][f] + h[n][f]) / cnt[n]   (self-loop folded in)
__global__ void sage_finalize_kernel(const float* __restrict__ sum,
                                     const float* __restrict__ hself,
                                     const float* __restrict__ cnt,
                                     float* __restrict__ agg, int total) {
    int i = blockIdx.x * blockDim.x + threadIdx.x;
    if (i >= total) return;
    int n = i >> 7;  // / 128
    agg[i] = (sum[i] + hself[i]) * (1.0f / cnt[n]);
}

// Row-wise L2 normalize, 64 cols: one wave per row, lane holds 2 floats.
__global__ void sage_l2norm_kernel(float* __restrict__ out, int N) {
    int t    = blockIdx.x * blockDim.x + threadIdx.x;
    int row  = t >> 5;
    int lane = t & 31;
    if (row >= N) return;
    float2* p = reinterpret_cast<float2*>(out + (size_t)row * 64);
    float2 v = p[lane];
    float ss = v.x * v.x + v.y * v.y;
    #pragma unroll
    for (int off = 16; off > 0; off >>= 1) ss += __shfl_xor(ss, off, 32);
    float scale = 1.0f / fmaxf(sqrtf(ss), 1e-12f);
    v.x *= scale;
    v.y *= scale;
    p[lane] = v;
}

// ---------------- WMMA GEMM: out[M,NC] = A[M,128] @ W[128,NC] + bias ----------------
// One 16-row M-tile per block; wave w computes N-tile [16w, 16w+16).
// A tile staged in LDS as f16 (two b128 ds_loads per lane per K-step);
// W gathered from global (hot in L2) with inline f32->f16 convert.
//
// Fragment layouts (ISA 7.12.2, wave32):
//   A 16x32 f16 : lane (g=lane>>4, m=lane&15): halves 0..7  -> A[m][32k + 8g + i]
//                                              halves 8..15 -> A[m][32k + 16 + 8g + i]
//   B 32x16 f16 : lane (g, n=lane&15):         halves 0..7  -> B[32k + 8g + i][n]
//                                              halves 8..15 -> B[32k + 16 + 8g + i][n]
//   C/D 16x16 f32: c[r] = D[8g + r][n]
template <int NC, bool RELU>
__global__ void sage_gemm_wmma_kernel(const float* __restrict__ A,
                                      const float* __restrict__ W,
                                      const float* __restrict__ bias,
                                      float* __restrict__ out, int M) {
    __shared__ _Float16 As[16 * DK];  // 4 KB

    const int m0 = blockIdx.x * 16;

    // Cooperative load + convert of the 16x128 A tile (clamped rows for safety).
    for (int i = threadIdx.x; i < 16 * DK; i += blockDim.x) {
        int r = i >> 7;
        int k = i & (DK - 1);
        int row = m0 + r;
        if (row >= M) row = M - 1;
        As[i] = (_Float16)A[(size_t)row * DK + k];
    }
    __syncthreads();

    const int wave = threadIdx.x >> 5;
    const int lane = threadIdx.x & 31;
    const int g    = lane >> 4;   // lane group (0/1)
    const int hl   = lane & 15;   // row (A) / col (B) within tile
    const int n2   = wave * 16 + hl;

    v8f c = {};

    #pragma unroll
    for (int ks = 0; ks < DK / 32; ++ks) {
        const int k0 = ks * 32;

        // A fragment: two aligned 16B LDS loads.
        v8h lo = *reinterpret_cast<const v8h*>(&As[(hl << 7) + k0 + (g << 3)]);
        v8h hi = *reinterpret_cast<const v8h*>(&As[(hl << 7) + k0 + 16 + (g << 3)]);
        v16h a;
        #pragma unroll
        for (int i = 0; i < 8; ++i) { a[i] = lo[i]; a[8 + i] = hi[i]; }

        // B fragment: strided gathers of W (lanes 0..15 are consecutive floats).
        v16h b;
        #pragma unroll
        for (int i = 0; i < 8; ++i) {
            b[i]     = (_Float16)W[(size_t)(k0 + (g << 3) + i) * NC + n2];
            b[8 + i] = (_Float16)W[(size_t)(k0 + 16 + (g << 3) + i) * NC + n2];
        }

        // (neg_a, A, neg_b, B, c_mod, C, reuse_a, reuse_b)
        c = __builtin_amdgcn_wmma_f32_16x16x32_f16(false, a, false, b,
                                                   (short)0, c, false, false);
    }

    const float bv = bias[n2];
    #pragma unroll
    for (int r = 0; r < 8; ++r) {
        int row = m0 + (g << 3) + r;
        if (row < M) {
            float v = c[r] + bv;
            if (RELU) v = fmaxf(v, 0.0f);
            out[(size_t)row * NC + n2] = v;
        }
    }
}

// ---------------------------------------------------------------------------

extern "C" void kernel_launch(void* const* d_in, const int* in_sizes, int n_in,
                              void* d_out, int out_size, void* d_ws, size_t ws_size,
                              hipStream_t stream) {
    const float* x     = (const float*)d_in[0];
    const int*   ei    = (const int*)  d_in[1];
    const float* W_pre = (const float*)d_in[2];
    const float* b_pre = (const float*)d_in[3];
    const float* W1    = (const float*)d_in[4];
    const float* b1    = (const float*)d_in[5];
    const float* W2    = (const float*)d_in[6];
    const float* b2    = (const float*)d_in[7];

    const int N = in_sizes[0] / DK;   // 50000
    const int E = in_sizes[1] / 2;    // 800000
    const int* src = ei;
    const int* dst = ei + E;

    // Workspace layout (all f32): h | sum | agg | cnt  (~77 MB)
    float* h    = (float*)d_ws;
    float* sum  = h   + (size_t)N * DK;
    float* agg  = sum + (size_t)N * DK;
    float* cnt  = agg + (size_t)N * DK;
    float* outp = (float*)d_out;

    const int TPB     = 256;
    const int mtiles  = (N + 15) / 16;
    const int total   = N * DK;
    const int gTotal  = (total + TPB - 1) / TPB;
    const int gN      = (N + TPB - 1) / TPB;
    const int gE      = (E + TPB - 1) / TPB;
    const int gScat   = (E * 32 + TPB - 1) / TPB;   // wave per edge
    const int gNorm   = (N * 32 + TPB - 1) / TPB;   // wave per row

    // 1) h = x @ W_pre + b_pre
    sage_gemm_wmma_kernel<DK, false><<<mtiles, 256, 0, stream>>>(x, W_pre, b_pre, h, N);

    // 2) degrees (self-loop -> init 1.0)
    sage_fill_kernel<<<gN, TPB, 0, stream>>>(cnt, 1.0f, N);
    sage_degree_kernel<<<gE, TPB, 0, stream>>>(dst, cnt, E);

    // 3) conv1 aggregation
    sage_fill_kernel<<<gTotal, TPB, 0, stream>>>(sum, 0.0f, total);
    sage_scatter_kernel<<<gScat, TPB, 0, stream>>>(h, src, dst, sum, E);
    sage_finalize_kernel<<<gTotal, TPB, 0, stream>>>(sum, h, cnt, agg, total);

    // 4) h2 = relu(agg @ W1 + b1)   (h2 reuses the h buffer)
    sage_gemm_wmma_kernel<DK, true><<<mtiles, 256, 0, stream>>>(agg, W1, b1, h, N);

    // 5) conv2 aggregation
    sage_fill_kernel<<<gTotal, TPB, 0, stream>>>(sum, 0.0f, total);
    sage_scatter_kernel<<<gScat, TPB, 0, stream>>>(h, src, dst, sum, E);
    sage_finalize_kernel<<<gTotal, TPB, 0, stream>>>(sum, h, cnt, agg, total);

    // 6) out = agg @ W2 + b2  (NC = 64 -> 4 waves / block), then L2-normalize rows
    sage_gemm_wmma_kernel<64, false><<<mtiles, 128, 0, stream>>>(agg, W2, b2, outp, N);
    sage_l2norm_kernel<<<gNorm, TPB, 0, stream>>>(outp, N);
}
